// SubRegions_26199300505718
// MI455X (gfx1250) — matrix operand order repaired and checked by reading, not compiled, not run on hardware
//
#include <hip/hip_runtime.h>

#define B_      2
#define C_      256
#define O_      256
#define H_      128
#define W_      128
#define HW_     (H_*W_)
#define HP_     130
#define PLANE_  (HP_*HP_)
#define CK_     2304          // 9 taps * 256 channels, tap-major
#define NSTEP_  (CK_/32)      // 72 K-steps of 32
#define GROUPS_ 32
#define GC_     (O_/GROUPS_)  // 8 channels per group
#define EPS_    1e-5f

typedef __attribute__((ext_vector_type(16))) __bf16        bf16x16;
typedef __attribute__((ext_vector_type(8)))  float         f32x8;
typedef __attribute__((ext_vector_type(4)))  unsigned int  u32x4;

union Frag { u32x4 u[2]; bf16x16 v; };

__device__ __forceinline__ unsigned short f2bf(float f) {
    unsigned int u = __float_as_uint(f);
    u += 0x7FFFu + ((u >> 16) & 1u);          // round to nearest even
    return (unsigned short)(u >> 16);
}

// ---------------- 1. zero-pad x -> xp[B,C,130,130] ----------------
__global__ void pad_kernel(const float* __restrict__ x, float* __restrict__ xp, int total) {
    int i = blockIdx.x * 256 + threadIdx.x;
    if (i >= total) return;
    int pw = i % HP_;  int t  = i / HP_;
    int ph = t % HP_;  int bc = t / HP_;
    float v = 0.f;
    if (ph >= 1 && ph <= H_ && pw >= 1 && pw <= W_)
        v = x[(size_t)bc * HW_ + (ph - 1) * W_ + (pw - 1)];
    xp[(size_t)bc * PLANE_ + ph * HP_ + pw] = v;
}

// ------------- 2. weights -> bf16, [O][tap*256 + c] layout -------------
__global__ void wcvt_kernel(const float* __restrict__ w, unsigned short* __restrict__ wbf) {
    int i = blockIdx.x * 256 + threadIdx.x;
    if (i >= O_ * CK_) return;
    int o = i / CK_, ck = i % CK_;
    int t = ck / C_, c = ck % C_;
    wbf[i] = f2bf(w[(size_t)o * CK_ + c * 9 + t]);   // w is [O][C][3][3]
}

// ------- 3. offset conv + bilinear tap metadata (idx4 + weight4) -------
__global__ __launch_bounds__(256) void offmeta_kernel(
    const float* __restrict__ xp, const float* __restrict__ w_off,
    const float* __restrict__ b_off,
    int4* __restrict__ midx, float4* __restrict__ mwt)
{
    int p = blockIdx.x * 256 + threadIdx.x;
    if (p >= B_ * HW_) return;
    int b = p / HW_, hw = p % HW_;
    int h = hw / W_, w = hw % W_;

    float acc[18];
    #pragma unroll
    for (int i = 0; i < 18; ++i) acc[i] = 0.f;

    const float* xb = xp + (size_t)b * C_ * PLANE_ + h * HP_ + w;  // 3x3 window top-left
    for (int c = 0; c < C_; ++c) {
        const float* xc = xb + (size_t)c * PLANE_;
        float xv[9];
        #pragma unroll
        for (int ky = 0; ky < 3; ++ky)
            #pragma unroll
            for (int kx = 0; kx < 3; ++kx)
                xv[ky * 3 + kx] = xc[ky * HP_ + kx];
        const float* wc = w_off + c * 9;
        #pragma unroll
        for (int oc = 0; oc < 18; ++oc) {
            const float* wr = wc + (size_t)oc * CK_;
            float s = 0.f;
            #pragma unroll
            for (int t = 0; t < 9; ++t) s += xv[t] * wr[t];
            acc[oc] += s;
        }
    }

    #pragma unroll
    for (int t = 0; t < 9; ++t) {
        float py = acc[t]     + b_off[t]     + (float)(t / 3) + (float)h;
        float px = acc[9 + t] + b_off[9 + t] + (float)(t % 3) + (float)w;
        py = fminf(fmaxf(py, 0.f), (float)(HP_ - 1));
        px = fminf(fmaxf(px, 0.f), (float)(HP_ - 1));
        float y0f = floorf(py), x0f = floorf(px);
        float y1f = fminf(y0f + 1.f, (float)(HP_ - 1));
        float x1f = fminf(x0f + 1.f, (float)(HP_ - 1));
        float wy = py - y0f, wx = px - x0f;
        int y0 = (int)y0f, x0 = (int)x0f, y1 = (int)y1f, x1 = (int)x1f;
        size_t m = ((size_t)b * 9 + t) * HW_ + hw;
        midx[m] = make_int4(y0 * HP_ + x0, y0 * HP_ + x1, y1 * HP_ + x0, y1 * HP_ + x1);
        mwt[m]  = make_float4((1.f - wy) * (1.f - wx), (1.f - wy) * wx,
                              wy * (1.f - wx),         wy * wx);
    }
}

// ------- 4. fused deformable implicit-GEMM, bf16 WMMA, f32 acc -------
// Block tile: O=256 x N=64 pixels. 8 waves; each wave: 2 M-tiles x 4 N-tiles.
// Software-pipelined: gather for step s+1 overlaps WMMAs of step s
// (double-buffered LDS B-tile, one barrier per K-step).
__global__ __launch_bounds__(256) void dconv_gemm_kernel(
    const float* __restrict__ xp,
    const unsigned short* __restrict__ wbf,
    const int4* __restrict__ midx,
    const float4* __restrict__ mwt,
    const float* __restrict__ bias,
    float* __restrict__ y)
{
    // B-tile staged in wave32 B-fragment layout: [buf][ntile(4)][lane(32)][k(16)]
    __shared__ __align__(16) unsigned short sV[2][4 * 32 * 16];

    const int tid  = threadIdx.x;
    const int lane = tid & 31;
    const int wv   = tid >> 5;
    const int n0   = blockIdx.x * 64;

    // gather role: this thread produces k = gk0..gk0+7 for pixel n0+gn
    const int gn  = tid & 63;
    const int gk0 = (tid >> 6) << 3;           // 0,8,16,24
    const int p   = n0 + gn;
    const int gb  = p / HW_;
    const int ghw = p % HW_;
    const int gnt = gn >> 4;                   // n-tile 0..3
    const int gln = gn & 15;
    const float* xbase = xp + (size_t)gb * C_ * PLANE_;
    const size_t mb0   = (size_t)gb * 9 * HW_ + ghw;

    // compute role
    const int mo = wv * 32;                    // this wave's O base

    f32x8 acc[2][4];
    #pragma unroll
    for (int mt = 0; mt < 2; ++mt)
        #pragma unroll
        for (int nt = 0; nt < 4; ++nt)
            #pragma unroll
            for (int r = 0; r < 8; ++r) acc[mt][nt][r] = 0.f;

    // tap metadata for the step currently being gathered
    int4   mi  = midx[mb0];
    float4 mw4 = mwt[mb0];

    // gather K-step `step` (32 c-values x 64 pixels) into buffer `buf`
    auto gather = [&](int step, int buf) {
        const int c0 = (step & 7) << 5;
        #pragma unroll
        for (int j = 0; j < 8; ++j) {
            const int kk = gk0 + j;            // 0..31 within K-step
            const float* xc = xbase + (size_t)(c0 + kk) * PLANE_;
            float v = fmaf(mw4.x, xc[mi.x],
                      fmaf(mw4.y, xc[mi.y],
                      fmaf(mw4.z, xc[mi.z], mw4.w * xc[mi.w])));
            // B layout: lanes 0-15 hold K 0-15, lanes 16-31 hold K 16-31
            const int bl = gln + (kk & 16);
            sV[buf][((gnt * 32 + bl) << 4) + (kk & 15)] = f2bf(v);
        }
    };

    gather(0, 0);
    __syncthreads();

    for (int step = 0; step < NSTEP_; ++step) {
        // ---- prefetch-gather next step into the other buffer ----
        const int nxt = step + 1;
        if (nxt < NSTEP_) {
            if ((nxt & 7) == 0) {              // uniform branch: new tap
                const size_t m = mb0 + (size_t)(nxt >> 3) * HW_;
                mi  = midx[m];
                mw4 = mwt[m];
            }
            gather(nxt, nxt & 1);
        }

        // ---- WMMAs for current step from buffer step&1 ----
        const int k0 = step << 5;              // tap-major: ck == step*32 + kk
        // A fragments: lane<16 -> K 0-15; lane>=16 -> K 16-31
        Frag a[2];
        #pragma unroll
        for (int mt = 0; mt < 2; ++mt) {
            const int o  = mo + mt * 16 + (lane & 15);
            const int kb = k0 + ((lane & 16) ? 16 : 0);
            const unsigned short* wr = wbf + (size_t)o * CK_ + kb;
            a[mt].u[0] = *(const u32x4*)(wr);
            a[mt].u[1] = *(const u32x4*)(wr + 8);
        }
        Frag bfr[4];
        #pragma unroll
        for (int nt = 0; nt < 4; ++nt) {
            const unsigned short* s = &sV[step & 1][(nt * 32 + lane) << 4];
            bfr[nt].u[0] = *(const u32x4*)(s);
            bfr[nt].u[1] = *(const u32x4*)(s + 8);
        }
        #pragma unroll
        for (int mt = 0; mt < 2; ++mt)
            #pragma unroll
            for (int nt = 0; nt < 4; ++nt)
                acc[mt][nt] = __builtin_amdgcn_wmma_f32_16x16x32_bf16(
                    false, a[mt].v, false, bfr[nt].v,
                    (short)0, acc[mt][nt], false, false);

        __syncthreads();   // next-step gathers visible; this buffer free in 2 steps
    }

    // epilogue: C/D layout -> lanes 0-15: M=r, N=lane; lanes 16-31: M=8+r
    #pragma unroll
    for (int mt = 0; mt < 2; ++mt) {
        #pragma unroll
        for (int nt = 0; nt < 4; ++nt) {
            const int N  = lane & 15;
            const int pp = n0 + nt * 16 + N;
            const int bb = pp / HW_;
            const int hh = pp % HW_;
            #pragma unroll
            for (int r = 0; r < 8; ++r) {
                const int M = ((lane & 16) ? 8 : 0) + r;
                const int o = mo + mt * 16 + M;
                y[((size_t)bb * O_ + o) * HW_ + hh] = acc[mt][nt][r] + bias[o];
            }
        }
    }
}

// ---------------- 5. GroupNorm stats (mean, rsigma) per (b,g) ----------------
__global__ __launch_bounds__(256) void gn_stats_kernel(
    const float* __restrict__ y, float2* __restrict__ stats)
{
    const int bg = blockIdx.x;                 // 0..63
    const int b = bg / GROUPS_, g = bg % GROUPS_;
    const float* base = y + ((size_t)b * O_ + g * GC_) * HW_;  // contiguous chunk
    float s = 0.f, s2 = 0.f;
    for (int i = threadIdx.x; i < GC_ * HW_; i += 256) {
        float v = base[i];
        s += v; s2 += v * v;
    }
    __shared__ float rs[256], rs2[256];
    rs[threadIdx.x] = s; rs2[threadIdx.x] = s2;
    __syncthreads();
    for (int off = 128; off > 0; off >>= 1) {
        if (threadIdx.x < off) {
            rs[threadIdx.x]  += rs[threadIdx.x + off];
            rs2[threadIdx.x] += rs2[threadIdx.x + off];
        }
        __syncthreads();
    }
    if (threadIdx.x == 0) {
        float n   = (float)(GC_ * HW_);
        float mu  = rs[0] / n;
        float var = rs2[0] / n - mu * mu;
        stats[bg] = make_float2(mu, rsqrtf(var + EPS_));
    }
}

// ---------------- 6. GroupNorm apply ----------------
__global__ void gn_apply_kernel(const float* __restrict__ y,
                                const float2* __restrict__ stats,
                                const float* __restrict__ gnw,
                                const float* __restrict__ gnb,
                                float* __restrict__ out)
{
    size_t i = (size_t)blockIdx.x * 256 + threadIdx.x;
    if (i >= (size_t)B_ * O_ * HW_) return;
    int o = (int)((i / HW_) % O_);
    int b = (int)(i / ((size_t)O_ * HW_));
    float2 st = stats[b * GROUPS_ + o / GC_];
    out[i] = (y[i] - st.x) * st.y * gnw[o] + gnb[o];
}

extern "C" void kernel_launch(void* const* d_in, const int* in_sizes, int n_in,
                              void* d_out, int out_size, void* d_ws, size_t ws_size,
                              hipStream_t stream) {
    (void)in_sizes; (void)n_in; (void)out_size; (void)ws_size;
    const float* x     = (const float*)d_in[0];
    const float* w_off = (const float*)d_in[1];
    const float* b_off = (const float*)d_in[2];
    const float* w     = (const float*)d_in[3];
    const float* bias  = (const float*)d_in[4];
    const float* gnw   = (const float*)d_in[5];
    const float* gnb   = (const float*)d_in[6];
    float* out = (float*)d_out;

    char* ws = (char*)d_ws;
    size_t off = 0;
    auto alloc = [&](size_t bytes) -> void* {
        void* ptr = ws + off;
        off += (bytes + 255) & ~(size_t)255;
        return ptr;
    };
    float*          xp    = (float*)         alloc((size_t)B_ * C_ * PLANE_ * 4);  // 34.6 MB
    unsigned short* wbf   = (unsigned short*)alloc((size_t)O_ * CK_ * 2);          // 1.2 MB
    int4*           midx  = (int4*)          alloc((size_t)B_ * 9 * HW_ * 16);     // 4.7 MB
    float4*         mwt   = (float4*)        alloc((size_t)B_ * 9 * HW_ * 16);     // 4.7 MB
    float*          ybuf  = (float*)         alloc((size_t)B_ * O_ * HW_ * 4);     // 33.5 MB
    float2*         stats = (float2*)        alloc(B_ * GROUPS_ * 8);

    pad_kernel<<<(B_ * C_ * PLANE_ + 255) / 256, 256, 0, stream>>>(x, xp, B_ * C_ * PLANE_);
    wcvt_kernel<<<(O_ * CK_ + 255) / 256, 256, 0, stream>>>(w, wbf);
    offmeta_kernel<<<(B_ * HW_ + 255) / 256, 256, 0, stream>>>(xp, w_off, b_off, midx, mwt);
    dconv_gemm_kernel<<<(B_ * HW_) / 64, 256, 0, stream>>>(xp, wbf, midx, mwt, bias, ybuf);
    gn_stats_kernel<<<B_ * GROUPS_, 256, 0, stream>>>(ybuf, stats);
    gn_apply_kernel<<<(int)(((size_t)B_ * O_ * HW_ + 255) / 256), 256, 0, stream>>>(
        ybuf, stats, gnw, gnb, out);
}
